// VisionMambaBlock_90778428768287
// MI455X (gfx1250) — compile-verified, matrix operand order checked by reading, not run on hardware
//
#include <hip/hip_runtime.h>

typedef __bf16 bf16;
typedef __attribute__((ext_vector_type(16))) __bf16 v16bf;
typedef __attribute__((ext_vector_type(8)))  float  v8f;
typedef __attribute__((ext_vector_type(4)))  unsigned int v4u;

union FragU { v4u u[2]; v16bf v; };

// force global (addrspace 1) b128 loads, tracked by LOADcnt only
typedef const __attribute__((address_space(1))) v4u* gv4u_p;

__device__ __forceinline__ FragU load_frag(const bf16* base, int koff) {
    FragU f;
    const bf16* p = base + koff;
    f.u[0] = *(gv4u_p)(p);
    f.u[1] = *(gv4u_p)(p + 16);
    return f;
}

__device__ __forceinline__ float bf2f(bf16 b) {
    union { unsigned short s; bf16 b; } i; i.b = b;
    union { unsigned u; float f; } o; o.u = ((unsigned)i.s) << 16;
    return o.f;
}
__device__ __forceinline__ bf16 f2bf(float f) {
    union { float f; unsigned u; } a; a.f = f;
    unsigned r = a.u + 0x7FFFu + ((a.u >> 16) & 1u);  // round-to-nearest-even
    union { unsigned short s; bf16 b; } o; o.s = (unsigned short)(r >> 16);
    return o.b;
}

// ---------------------------------------------------------------------------
// Generic bf16 WMMA GEMM: C[M,N] = A[M,K] * W[N,K]^T
//   A rows optionally sequence-reversed on read (revA) or on store (revC),
//   per batch of length L (row = b*L + l  ->  b*L + (L-1-l)).
//   Output either fp32 (Cf) or bf16 (Cb) at column offset coff, row stride ldc.
// Wave tile: 16(M) x 64(N), block: 8 waves -> 128 x 64.
// Branchless inner loop (OOB columns clamped, masked at store), software-
// pipelined one K-step ahead; fragment loads forced to global_load_b128.
// ---------------------------------------------------------------------------
__global__ __launch_bounds__(256)
void gemm_bf16_wmma(const bf16* __restrict__ A, const bf16* __restrict__ W,
                    float* __restrict__ Cf, bf16* __restrict__ Cb,
                    int M, int N, int K, int ldc, int coff,
                    int revA, int revC, int L)
{
    const int wave = threadIdx.x >> 5;
    const int lane = threadIdx.x & 31;
    const int half = lane >> 4;       // 0: lanes 0-15, 1: lanes 16-31
    const int i16  = lane & 15;
    const int m0   = (blockIdx.x * 8 + wave) * 16;
    const int n0   = blockIdx.y * 64;
    if (m0 >= M) return;

    int rowA = m0 + i16;
    if (revA) { int b = rowA / L; int l = rowA - b * L; rowA = b * L + (L - 1 - l); }
    const bf16* pa = A + (size_t)rowA * K + half * 8;   // fixed base, k0-indexed

    // Clamp OOB columns once; their results are discarded by the store mask.
    bool nok[4];
    const bf16* pb[4];                                   // fixed bases
#pragma unroll
    for (int t = 0; t < 4; ++t) {
        int n = n0 + t * 16 + i16;
        nok[t] = (n < N);
        int nc = nok[t] ? n : (N - 1);
        pb[t] = W + (size_t)nc * K + half * 8;
    }

    v8f zero8 = {0.f,0.f,0.f,0.f,0.f,0.f,0.f,0.f};
    v8f acc[4];
#pragma unroll
    for (int t = 0; t < 4; ++t) acc[t] = zero8;

    // --- prologue: load K-step 0 fragments ---
    FragU a_cur = load_frag(pa, 0);
    FragU b_cur[4];
#pragma unroll
    for (int t = 0; t < 4; ++t) b_cur[t] = load_frag(pb[t], 0);

    // --- pipelined main loop: load k0+32 fragments, then WMMA on k0 ---
    for (int k0 = 0; k0 < K - 32; k0 += 32) {
        FragU a_n = load_frag(pa, k0 + 32);
        FragU b_n[4];
#pragma unroll
        for (int t = 0; t < 4; ++t) b_n[t] = load_frag(pb[t], k0 + 32);

        v16bf av = a_cur.v;
#pragma unroll
        for (int t = 0; t < 4; ++t)
            acc[t] = __builtin_amdgcn_wmma_f32_16x16x32_bf16(
                false, av, false, b_cur[t].v, (short)0, acc[t], false, false);
        a_cur = a_n;
#pragma unroll
        for (int t = 0; t < 4; ++t) b_cur[t] = b_n[t];
    }
    // --- epilogue: last K-step ---
    {
        v16bf av = a_cur.v;
#pragma unroll
        for (int t = 0; t < 4; ++t)
            acc[t] = __builtin_amdgcn_wmma_f32_16x16x32_bf16(
                false, av, false, b_cur[t].v, (short)0, acc[t], false, false);
    }

    // C/D layout: lane -> col n0+t*16+i16; VGPR r -> row m0 + r + 8*half
#pragma unroll
    for (int t = 0; t < 4; ++t) {
        if (!nok[t]) continue;
        int n = n0 + t * 16 + i16;
#pragma unroll
        for (int r = 0; r < 8; ++r) {
            int row = m0 + r + half * 8;
            int crow = row;
            if (revC) { int b = row / L; int l = row - b * L; crow = b * L + (L - 1 - l); }
            float v = acc[t][r];
            if (Cf) Cf[(size_t)crow * ldc + coff + n] = v;
            else    Cb[(size_t)crow * ldc + coff + n] = f2bf(v);
        }
    }
}

// ---------------------------------------------------------------------------
__global__ void cvt_f32_bf16(const float* __restrict__ in, bf16* __restrict__ out, size_t n)
{
    size_t i = (size_t)blockIdx.x * blockDim.x + threadIdx.x;
    if (i < n) out[i] = f2bf(in[i]);
}

__global__ __launch_bounds__(128)
void layernorm_bf16(const float* __restrict__ x, const float* __restrict__ g,
                    const float* __restrict__ be, bf16* __restrict__ xn, int D)
{
    __shared__ float ss[128], sq[128];
    const int row = blockIdx.x;
    const float* xr = x + (size_t)row * D;
    float s = 0.f, q = 0.f;
    for (int i = threadIdx.x; i < D; i += 128) { float v = xr[i]; s += v; q += v * v; }
    ss[threadIdx.x] = s; sq[threadIdx.x] = q;
    __syncthreads();
    for (int o = 64; o > 0; o >>= 1) {
        if ((int)threadIdx.x < o) { ss[threadIdx.x] += ss[threadIdx.x + o];
                                    sq[threadIdx.x] += sq[threadIdx.x + o]; }
        __syncthreads();
    }
    float mean = ss[0] / D;
    float var  = sq[0] / D - mean * mean;
    float rstd = rsqrtf(var + 1e-5f);
    for (int i = threadIdx.x; i < D; i += 128)
        xn[(size_t)row * D + i] = f2bf((xr[i] - mean) * rstd * g[i] + be[i]);
}

// depthwise causal conv (width 4, left pad 3) + SiLU, on first 768 cols of xz
__global__ void conv_silu(const bf16* __restrict__ xz, const float* __restrict__ cw,
                          const float* __restrict__ cb, bf16* __restrict__ xc,
                          int L, int Din, int ldz, size_t total)
{
    size_t idx = (size_t)blockIdx.x * blockDim.x + threadIdx.x;
    if (idx >= total) return;
    int d = (int)(idx % Din);
    size_t bl = idx / Din;          // b*L + l
    int l = (int)(bl % L);
    float a = cb[d];
#pragma unroll
    for (int j = 0; j < 4; ++j) {
        int ls = l - 3 + j;
        if (ls >= 0) a += cw[d * 4 + j] * bf2f(xz[(bl + (size_t)(ls - l)) * ldz + d]);
    }
    a = a / (1.f + expf(-a));       // SiLU
    xc[idx] = f2bf(a);
}

// Fused: dt = softplus(xp[:,:24]@dt_w[d]+dt_b), exp-recurrence over L,
// y += D*x, y *= silu(z).  One thread per (b, d) channel.
__global__ __launch_bounds__(256)
void selective_scan(const bf16* __restrict__ xc, const bf16* __restrict__ xz,
                    const float* __restrict__ xp, const float* __restrict__ dtw,
                    const float* __restrict__ dtb, const float* __restrict__ Alog,
                    const float* __restrict__ Dp, bf16* __restrict__ Y,
                    int Bn, int L, int ldz)
{
    int idx = blockIdx.x * blockDim.x + threadIdx.x;
    if (idx >= Bn * 768) return;
    int d = idx % 768;
    int b = idx / 768;
    float Ar[16], h[16];
#pragma unroll
    for (int n = 0; n < 16; ++n) { Ar[n] = -expf(Alog[d * 16 + n]); h[n] = 0.f; }
    float wt[24];
#pragma unroll
    for (int r = 0; r < 24; ++r) wt[r] = dtw[d * 24 + r];
    const float bias = dtb[d], Dv = Dp[d];

    for (int l = 0; l < L; ++l) {
        size_t m = (size_t)b * L + l;
        const float* xr = xp + m * 56;          // [dt_rank(24) | B(16) | C(16)]
        if (l + 1 < L) __builtin_prefetch(xr + 56, 0, 1);   // next step's row
        float dtv = bias;
#pragma unroll
        for (int r = 0; r < 24; ++r) dtv += wt[r] * xr[r];
        float dt = (dtv > 20.f) ? dtv : log1pf(expf(dtv));   // softplus
        float xt = bf2f(xc[m * 768 + d]);
        float dx = dt * xt;
        float y = 0.f;
#pragma unroll
        for (int n = 0; n < 16; ++n) {
            h[n] = expf(dt * Ar[n]) * h[n] + dx * xr[24 + n];
            y += h[n] * xr[40 + n];
        }
        y += xt * Dv;
        float z = bf2f(xz[m * (size_t)ldz + 768 + d]);
        y *= z / (1.f + expf(-z));              // * silu(z)
        Y[m * 768 + d] = f2bf(y);
    }
}

__global__ void gate_combine(const float* __restrict__ G, const float* __restrict__ gb,
                             const bf16* __restrict__ Comb, const float* __restrict__ x,
                             float* __restrict__ out, int Dm, size_t total)
{
    size_t idx = (size_t)blockIdx.x * blockDim.x + threadIdx.x;
    if (idx >= total) return;
    int c = (int)(idx % Dm);
    size_t row = idx / Dm;
    float g = G[idx] + gb[c];
    g = 1.f / (1.f + expf(-g));
    float yf = bf2f(Comb[row * 768 + c]);
    float yb = bf2f(Comb[row * 768 + 384 + c]);
    out[idx] = g * yf + (1.f - g) * yb + x[idx];
}

// ---------------------------------------------------------------------------
extern "C" void kernel_launch(void* const* d_in, const int* in_sizes, int n_in,
                              void* d_out, int out_size, void* d_ws, size_t ws_size,
                              hipStream_t stream)
{
    const int B = 32, L = 1024, Dm = 384, Din = 768, Dxz = 1536;
    const int M = B * L;                         // 32768 rows

    // setup_inputs() dict order: x, ln_gamma, ln_beta, gate_w, gate_b, f_*..., b_*...
    // (defensive: fall back to reference-signature order if sizes disagree)
    int idx_gw, idx_gb, base_f, base_b;
    if (in_sizes[3] == Dm * 2 * Dm) { idx_gw = 3;  idx_gb = 4;  base_f = 5;  base_b = 14; }
    else                            { base_f = 3;  base_b = 12; idx_gw = 21; idx_gb = 22; }

    const float* x      = (const float*)d_in[0];
    const float* lng    = (const float*)d_in[1];
    const float* lnb    = (const float*)d_in[2];
    const float* gate_w = (const float*)d_in[idx_gw];
    const float* gate_b = (const float*)d_in[idx_gb];

    // workspace carve-out (256B aligned blocks)
    char* ws = (char*)d_ws;
    size_t off = 0;
    auto alloc = [&](size_t bytes) -> void* {
        void* p = ws + off;
        off += (bytes + 255) & ~((size_t)255);
        return p;
    };
    bf16* w_in_f  = (bf16*)alloc((size_t)Dxz * Dm * 2);
    bf16* w_in_b  = (bf16*)alloc((size_t)Dxz * Dm * 2);
    bf16* w_xp_f  = (bf16*)alloc((size_t)56 * Din * 2);
    bf16* w_xp_b  = (bf16*)alloc((size_t)56 * Din * 2);
    bf16* w_out_f = (bf16*)alloc((size_t)Dm * Din * 2);
    bf16* w_out_b = (bf16*)alloc((size_t)Dm * Din * 2);
    bf16* w_gate  = (bf16*)alloc((size_t)Dm * Din * 2);
    bf16*  xn   = (bf16*) alloc((size_t)M * Dm  * 2);
    bf16*  xzb  = (bf16*) alloc((size_t)M * Dxz * 2);
    bf16*  xcb  = (bf16*) alloc((size_t)M * Din * 2);
    float* xp   = (float*)alloc((size_t)M * 56  * 4);
    bf16*  Yin  = (bf16*) alloc((size_t)M * Din * 2);
    bf16*  Comb = (bf16*) alloc((size_t)M * Din * 2);
    float* G    = (float*)alloc((size_t)M * Dm  * 4);

    auto cvt = [&](const void* src, bf16* dst, size_t n) {
        cvt_f32_bf16<<<dim3((unsigned)((n + 255) / 256)), dim3(256), 0, stream>>>(
            (const float*)src, dst, n);
    };
    cvt(d_in[base_f + 0], w_in_f,  (size_t)Dxz * Dm);
    cvt(d_in[base_b + 0], w_in_b,  (size_t)Dxz * Dm);
    cvt(d_in[base_f + 3], w_xp_f,  (size_t)56 * Din);
    cvt(d_in[base_b + 3], w_xp_b,  (size_t)56 * Din);
    cvt(d_in[base_f + 8], w_out_f, (size_t)Dm * Din);
    cvt(d_in[base_b + 8], w_out_b, (size_t)Dm * Din);
    cvt(gate_w,           w_gate,  (size_t)Dm * Din);

    layernorm_bf16<<<dim3(M), dim3(128), 0, stream>>>(x, lng, lnb, xn, Dm);

    auto gemm = [&](const bf16* A, const bf16* W, float* Cf, bf16* Cb,
                    int N, int K, int ldc, int coff, int revA, int revC) {
        dim3 grid(M / 128, (N + 63) / 64);
        gemm_bf16_wmma<<<grid, dim3(256), 0, stream>>>(
            A, W, Cf, Cb, M, N, K, ldc, coff, revA, revC, L);
    };

    for (int dir = 0; dir < 2; ++dir) {           // 0 = fwd, 1 = bwd (flipped seq)
        int bse = dir ? base_b : base_f;
        const float* cw  = (const float*)d_in[bse + 1];
        const float* cbv = (const float*)d_in[bse + 2];
        const float* dtw = (const float*)d_in[bse + 4];
        const float* dtb = (const float*)d_in[bse + 5];
        const float* Al  = (const float*)d_in[bse + 6];
        const float* Dp  = (const float*)d_in[bse + 7];
        bf16* w_in  = dir ? w_in_b  : w_in_f;
        bf16* w_xp  = dir ? w_xp_b  : w_xp_f;
        bf16* w_out = dir ? w_out_b : w_out_f;

        // in_proj: xz = (rev? flip(xn) : xn) @ in_w^T      (bf16 out)
        gemm(xn, w_in, nullptr, xzb, Dxz, Dm, Dxz, 0, dir, 0);
        // depthwise conv + SiLU -> xc
        size_t totc = (size_t)M * Din;
        conv_silu<<<dim3((unsigned)((totc + 255) / 256)), dim3(256), 0, stream>>>(
            xzb, cw, cbv, xcb, L, Din, Dxz, totc);
        // x_proj: xp = xc @ x_proj_w^T                      (fp32 out, N=56)
        gemm(xcb, w_xp, xp, nullptr, 56, Din, 56, 0, 0, 0);
        // fused dt + selective scan + D-skip + SiLU(z) gate -> Yin (bf16)
        int nscan = B * Din;
        selective_scan<<<dim3((nscan + 255) / 256), dim3(256), 0, stream>>>(
            xcb, xzb, xp, dtw, dtb, Al, Dp, Yin, B, L, Dxz);
        // out_proj -> Comb[:, dir*384 : dir*384+384], un-flip on store for bwd
        gemm(Yin, w_out, nullptr, Comb, Dm, Din, Din, dir ? Dm : 0, 0, dir);
    }

    // gate logits: G = [y_fwd | y_bwd] @ gate_w^T           (fp32 out)
    gemm(Comb, w_gate, G, nullptr, Dm, Din, Dm, 0, 0, 0);

    // y = sigmoid(G + b) * y_fwd + (1 - sigmoid) * y_bwd + residual
    size_t tot = (size_t)M * Dm;
    gate_combine<<<dim3((unsigned)((tot + 255) / 256)), dim3(256), 0, stream>>>(
        G, gate_b, Comb, x, (float*)d_out, Dm, tot);
}
